// Transformer_43198781063181
// MI455X (gfx1250) — matrix operand order repaired and checked by reading, not compiled, not run on hardware
//
#include <hip/hip_runtime.h>
#include <hip/hip_bf16.h>

// ---------------------------------------------------------------------------
// CDNA5 / gfx1250 transformer forward (compile-only target).
// GEMMs + attention on v_wmma_f32_16x16x32_bf16 (wave32), TDM for V-tiles.
// ---------------------------------------------------------------------------

typedef __attribute__((ext_vector_type(16))) __bf16 v16bf;
typedef __attribute__((ext_vector_type(8)))  __bf16 v8bf;
typedef __attribute__((ext_vector_type(8)))  float  v8f;
typedef __attribute__((ext_vector_type(4)))  float  f32x4;
typedef __attribute__((ext_vector_type(4)))  unsigned int u32x4;
typedef __attribute__((ext_vector_type(8)))  int    i32x8;
typedef __attribute__((ext_vector_type(4)))  int    i32x4;

#if __has_builtin(__builtin_amdgcn_tensor_load_to_lds)
#define HAVE_TDM 1
#else
#define HAVE_TDM 0
#endif

#define NTOK   16384      // 32 * 512
#define DMODEL 512
#define LEN    512
#define NHEAD  8
#define HDIM   64

#define SHUF16(lo, hi) __builtin_shufflevector((lo), (hi), 0,1,2,3,4,5,6,7,8,9,10,11,12,13,14,15)

// ===========================================================================
// Embedding + sinusoidal positional encoding.
// ===========================================================================
__global__ void embed_pe_kernel(const int* __restrict__ tok,
                                const float* __restrict__ emb,
                                float* __restrict__ out)
{
    const int p = blockIdx.x;        // token position
    const int c = threadIdx.x;       // 0..511
    const int l = p & (LEN - 1);
    const int t = tok[p];
    const float ang = (float)l * __expf(-((float)c * (1.0f / 512.0f)) * 9.210340371976184f);
    const float pe  = (c & 1) ? __cosf(ang) : __sinf(ang);
    out[(size_t)p * DMODEL + c] = emb[(size_t)t * DMODEL + c] + pe;
}

// ===========================================================================
// LayerNorm over D=512, one wave32 per row.
// ===========================================================================
__global__ __launch_bounds__(256)
void ln_kernel(const float* __restrict__ x,
               const float* __restrict__ g,
               const float* __restrict__ b,
               float* __restrict__ out, int nrow)
{
    const int wave = threadIdx.x >> 5;
    const int lane = threadIdx.x & 31;
    const int row  = blockIdx.x * 8 + wave;
    if (row >= nrow) return;
    const float* xr = x + (size_t)row * DMODEL;
    float vals[16];
    float s = 0.f, sq = 0.f;
#pragma unroll
    for (int i = 0; i < 16; ++i) {
        float v = xr[lane + 32 * i];
        vals[i] = v; s += v; sq += v * v;
    }
#pragma unroll
    for (int m = 16; m >= 1; m >>= 1) {
        s  += __shfl_xor(s,  m, 32);
        sq += __shfl_xor(sq, m, 32);
    }
    const float mean = s * (1.f / DMODEL);
    const float var  = sq * (1.f / DMODEL) - mean * mean;
    const float inv  = rsqrtf(var + 1e-5f);
    float* orow = out + (size_t)row * DMODEL;
#pragma unroll
    for (int i = 0; i < 16; ++i) {
        const int c = lane + 32 * i;
        orow[c] = (vals[i] - mean) * inv * g[c] + b[c];
    }
}

// ===========================================================================
// GEMM: C[M,N] = A[M,K](fp32->bf16) x W[K,N](fp32->bf16) + bias
// Template-specialized epilogue: BOUND (N guard), RELU, RES(idual), OUTF/OUTB.
// Requires M % 128 == 0, K % 32 == 0 (true for this model).
// Block 256 = 8 waves, tile 128x128, K-step 32, wave = 2x4 WMMA tiles.
// LDS rows padded to 40 bf16 (80B = 5x16B): aligned b128 ds ops, 16 distinct
// banks across 16 consecutive rows.
// ===========================================================================
template <bool BOUND, bool RELU, bool RES, bool OUTF, bool OUTB>
__global__ __launch_bounds__(256)
void gemm_kernel(const float* __restrict__ A, int lda,
                 const float* __restrict__ W,        // [K,N] row-major
                 const float* __restrict__ bias,     // [N]
                 const float* __restrict__ residual, // [M,N] (RES)
                 float* __restrict__ outF,
                 __bf16* __restrict__ outB,
                 int M, int N, int K)
{
    __shared__ __attribute__((aligned(16))) __bf16 As[128][40];
    __shared__ __attribute__((aligned(16))) __bf16 Bs[128][40];

    const int tid  = threadIdx.x;
    const int lane = tid & 31;
    const int wave = tid >> 5;
    const int bm   = blockIdx.y * 128;
    const int bn   = blockIdx.x * 128;
    const int wm   = (wave >> 1) * 32;   // 0,32,64,96
    const int wn   = (wave & 1) * 64;    // 0,64

    const int lrow = lane & 15;
    const int hi   = lane >> 4;          // 0 or 1

    v8f acc[2][4] = {};

    const int rstage = tid >> 1;         // 0..127
    const int kstage = (tid & 1) * 16;   // 0 or 16

    for (int k0 = 0; k0 < K; k0 += 32) {
        // ---- stage A tile: 16 contiguous fp32 -> 16 bf16, two b128 stores
        {
            const f32x4* src = (const f32x4*)(A + (size_t)(bm + rstage) * lda + k0 + kstage);
            __builtin_prefetch((const float*)src + 32, 0, 3);
            const f32x4 f0 = src[0], f1 = src[1], f2 = src[2], f3 = src[3];
            v8bf p0, p1;
#pragma unroll
            for (int e = 0; e < 4; ++e) {
                p0[e]     = (__bf16)f0[e];
                p0[4 + e] = (__bf16)f1[e];
                p1[e]     = (__bf16)f2[e];
                p1[4 + e] = (__bf16)f3[e];
            }
            *(v8bf*)&As[rstage][kstage]     = p0;
            *(v8bf*)&As[rstage][kstage + 8] = p1;
        }
        // ---- stage B tile transposed: Bs[col][k]
        {
            const int gc = bn + rstage;
            const float* src = W + (size_t)(k0 + kstage) * N + gc;
            v8bf q0v = {}, q1v = {};
            if (!BOUND || gc < N) {
                __builtin_prefetch(src + (size_t)32 * N, 0, 3);
#pragma unroll
                for (int e = 0; e < 8; ++e) q0v[e] = (__bf16)src[(size_t)e * N];
#pragma unroll
                for (int e = 0; e < 8; ++e) q1v[e] = (__bf16)src[(size_t)(8 + e) * N];
            }
            *(v8bf*)&Bs[rstage][kstage]     = q0v;
            *(v8bf*)&Bs[rstage][kstage + 8] = q1v;
        }
        __syncthreads();

        // ---- fragments (aligned b128 LDS loads) ----
        v16bf afrag[2], bfrag[4];
#pragma unroll
        for (int mi = 0; mi < 2; ++mi) {
            const v8bf* pa = (const v8bf*)&As[wm + mi * 16 + lrow][0];
            afrag[mi] = SHUF16(pa[hi], pa[2 + hi]);          // K: hi*8.. and 16+hi*8..
        }
#pragma unroll
        for (int ni = 0; ni < 4; ++ni) {
            const v8bf* pb = (const v8bf*)&Bs[wn + ni * 16 + lrow][0];
            bfrag[ni] = SHUF16(pb[hi], pb[2 + hi]);          // K: hi*8.. and 16+hi*8..
        }

#pragma unroll
        for (int mi = 0; mi < 2; ++mi)
#pragma unroll
            for (int ni = 0; ni < 4; ++ni)
                acc[mi][ni] = __builtin_amdgcn_wmma_f32_16x16x32_bf16(
                    false, afrag[mi], false, bfrag[ni],
                    (short)0, acc[mi][ni], false, false);
        __syncthreads();
    }

    // ---- epilogue (fully branch-free per specialization) ----
#pragma unroll
    for (int ni = 0; ni < 4; ++ni) {
        const int gc = bn + wn + ni * 16 + lrow;
        if (BOUND && gc >= N) continue;
#pragma unroll
        for (int mi = 0; mi < 2; ++mi) {
#pragma unroll
            for (int r = 0; r < 8; ++r) {
                const int gr = bm + wm + mi * 16 + r + 8 * hi;
                const size_t o = (size_t)gr * N + gc;
                float v = acc[mi][ni][r] + bias[gc];
                if (RES)  v += residual[o];
                if (RELU) v = v > 0.f ? v : 0.f;
                if (OUTF) outF[o] = v;
                if (OUTB) outB[o] = (__bf16)v;
            }
        }
    }
}

// ===========================================================================
// Fused attention, one (b,h) x 16 query rows per block (256 thr = 8 waves).
//   phase1: S = Q K^T / 8        (8 waves x 64 keys, WMMA) -> LDS fp32
//   phase2: row softmax          (unnormalized; 1/sum folded into epilogue)
//   phase3: O = P V              (4 waves x 16 dims, WMMA) -> fp32
// V block (512x64 bf16, 64KB) is DMA'd into LDS by the Tensor Data Mover
// while phase1 runs; s_wait_tensorcnt before the barrier into phase2.
// ===========================================================================
__global__ __launch_bounds__(256)
void attn_kernel(const __bf16* __restrict__ Q,
                 const __bf16* __restrict__ Km,
                 const __bf16* __restrict__ V,
                 float* __restrict__ O,
                 int causal)
{
    __shared__ float  S[16][512 + 4];
    __shared__ __attribute__((aligned(16))) __bf16 P[16][512 + 8];
    __shared__ float  invsum[16];
#if HAVE_TDM
    __shared__ __attribute__((aligned(16))) __bf16 Vs[512][64];
#endif

    const int tid  = threadIdx.x;
    const int lane = tid & 31;
    const int wave = tid >> 5;
    const int bh   = blockIdx.y;              // b*8+h
    const int b    = bh >> 3, h = bh & 7;
    const int q0   = blockIdx.x * 16;
    const size_t base = (size_t)b * LEN * DMODEL + (size_t)h * HDIM;

    const int lrow = lane & 15;
    const int hi   = lane >> 4;

#if HAVE_TDM
    // ---- kick off TDM: V block [512 keys][64 dims] bf16 -> LDS Vs --------
    if (wave == 0) {
        const unsigned int lds = (unsigned int)(uintptr_t)&Vs[0][0];
        const unsigned long long ga = (unsigned long long)(uintptr_t)(V + base);
        u32x4 g0;
        g0[0] = 1u;                                       // count=1
        g0[1] = lds;                                      // lds_addr
        g0[2] = (unsigned int)ga;                         // global_addr[31:0]
        g0[3] = (unsigned int)((ga >> 32) & 0x01FFFFFFu)  // global_addr[56:32]
              | (2u << 30);                               // type=2 (image)
        i32x8 g1;
        g1[0] = 0x00010000;            // workgroup_mask=0, data_size=1 (2B)
        g1[1] = (int)(64u << 16);      // tensor_dim0 = 64   [63:48]
        g1[2] = (int)(512u << 16);     // tensor_dim1 = 512  [95:80]
        g1[3] = (int)(64u << 16);      // tile_dim0   = 64   [127:112]
        g1[4] = 512;                   // tile_dim1   = 512  [143:128]
        g1[5] = 512;                   // tensor_dim0_stride = 512 [191:160]
        g1[6] = 0;
        g1[7] = 0;
        const i32x4 z4 = {0, 0, 0, 0};
#if __has_include(<hip/amd_detail/amd_gfx1250_TDM.h>)
        const i32x8 z8 = {0, 0, 0, 0, 0, 0, 0, 0};
        __builtin_amdgcn_tensor_load_to_lds(g0, g1, z4, z4, z8, 0);
#else
        __builtin_amdgcn_tensor_load_to_lds(g0, g1, z4, z4, 0);
#endif
    }
#endif

    // ---------------- phase 1: scores for keys [wave*64, wave*64+64) --------
    {
        v8f sc[4] = {};
#pragma unroll
        for (int kk = 0; kk < HDIM; kk += 32) {
            const v8bf* qp = (const v8bf*)(Q + base + (size_t)(q0 + lrow) * DMODEL + kk + hi * 8);
            const v16bf a = SHUF16(qp[0], qp[2]);
            v16bf bt[4];
#pragma unroll
            for (int t = 0; t < 4; ++t) {
                const int key = wave * 64 + t * 16 + lrow;
                const v8bf* kp = (const v8bf*)(Km + base + (size_t)key * DMODEL + kk + hi * 8);
                bt[t] = SHUF16(kp[0], kp[2]);
            }
#pragma unroll
            for (int t = 0; t < 4; ++t)
                sc[t] = __builtin_amdgcn_wmma_f32_16x16x32_bf16(
                    false, a, false, bt[t], (short)0, sc[t], false, false);
        }
#pragma unroll
        for (int t = 0; t < 4; ++t) {
#pragma unroll
            for (int r = 0; r < 8; ++r) {
                const int qi  = r + 8 * hi;
                const int key = wave * 64 + t * 16 + lrow;
                float v = sc[t][r] * 0.125f;                  // 1/sqrt(64)
                if (causal && key > q0 + qi) v = -1e30f;
                S[qi][key] = v;
            }
        }
    }
#if HAVE_TDM
    if (wave == 0) __builtin_amdgcn_s_wait_tensorcnt(0);
#endif
    __syncthreads();

    // ---------------- phase 2: softmax, 2 rows per wave ---------------------
    {
        const int row = wave * 2 + hi;      // 0..15
        const int i   = lrow;               // 16 lanes cooperate per row
        float mx = -1e30f;
        for (int j = i; j < 512; j += 16) mx = fmaxf(mx, S[row][j]);
#pragma unroll
        for (int m = 1; m < 16; m <<= 1) mx = fmaxf(mx, __shfl_xor(mx, m, 32));
        float sum = 0.f;
        for (int j = i; j < 512; j += 16) {
            const float e = __expf(S[row][j] - mx);
            sum += e;
            P[row][j] = (__bf16)e;
        }
#pragma unroll
        for (int m = 1; m < 16; m <<= 1) sum += __shfl_xor(sum, m, 32);
        if (i == 0) invsum[row] = 1.f / sum;
    }
    __syncthreads();

    // ---------------- phase 3: O = P x V, waves 0..3 ------------------------
    if (wave < 4) {
        v8f oc = {};
        const int d = wave * 16 + lrow;     // column within head
        for (int kk = 0; kk < 512; kk += 32) {
            const v8bf* pp = (const v8bf*)&P[lrow][kk + hi * 8];
            const v16bf a = SHUF16(pp[0], pp[2]);
            v16bf bf;
#pragma unroll
            for (int e = 0; e < 16; ++e) {
                const int key = kk + hi * 8 + (e & 7) + ((e >> 3) << 4);
#if HAVE_TDM
                bf[e] = Vs[key][d];
#else
                bf[e] = V[base + (size_t)key * DMODEL + d];
#endif
            }
            oc = __builtin_amdgcn_wmma_f32_16x16x32_bf16(
                false, a, false, bf, (short)0, oc, false, false);
        }
#pragma unroll
        for (int r = 0; r < 8; ++r) {
            const int qi = r + 8 * hi;
            O[base + (size_t)(q0 + qi) * DMODEL + wave * 16 + lrow] = oc[r] * invsum[qi];
        }
    }
}

// ===========================================================================
// Host side
// ===========================================================================
struct Lin { const float* w; const float* b; };
struct Mha { Lin k; const float* ln_b; const float* ln_g; Lin o; Lin q; Lin v; };
struct Fpn { Lin fc1; Lin fc2; const float* ln_b; const float* ln_g; };

static void launch_gemm(hipStream_t st, const float* A, int lda, const Lin& L,
                        const float* res, float* oF, __bf16* oB,
                        int M, int N, int K, int relu)
{
    dim3 g((N + 127) / 128, (M + 127) / 128);
    if (oB)
        gemm_kernel<false, false, false, false, true><<<g, 256, 0, st>>>(
            A, lda, L.w, L.b, nullptr, nullptr, oB, M, N, K);
    else if (relu)
        gemm_kernel<false, true, false, true, false><<<g, 256, 0, st>>>(
            A, lda, L.w, L.b, nullptr, oF, nullptr, M, N, K);
    else if (res)
        gemm_kernel<false, false, true, true, false><<<g, 256, 0, st>>>(
            A, lda, L.w, L.b, res, oF, nullptr, M, N, K);
    else  // classifier path: N=1000 needs bounds
        gemm_kernel<true, false, false, true, false><<<g, 256, 0, st>>>(
            A, lda, L.w, L.b, nullptr, oF, nullptr, M, N, K);
}

static void launch_ln(hipStream_t st, const float* x, const float* g,
                      const float* b, float* out)
{
    ln_kernel<<<NTOK / 8, 256, 0, st>>>(x, g, b, out, NTOK);
}

struct Ws {
    float *h, *d, *xln, *ob, *mid;
    __bf16 *qb, *kb, *vb;
};

static void run_mha(hipStream_t st, const Mha& m, float* qstream,
                    const float* kvstream, int causal, Ws& w)
{
    launch_ln(st, qstream, m.ln_g, m.ln_b, w.xln);
    launch_gemm(st, w.xln, DMODEL, m.q, nullptr, nullptr, w.qb, NTOK, DMODEL, DMODEL, 0);
    if (kvstream != qstream)
        launch_ln(st, kvstream, m.ln_g, m.ln_b, w.xln);
    launch_gemm(st, w.xln, DMODEL, m.k, nullptr, nullptr, w.kb, NTOK, DMODEL, DMODEL, 0);
    launch_gemm(st, w.xln, DMODEL, m.v, nullptr, nullptr, w.vb, NTOK, DMODEL, DMODEL, 0);
    attn_kernel<<<dim3(LEN / 16, 32 * NHEAD), 256, 0, st>>>(w.qb, w.kb, w.vb, w.ob, causal);
    launch_gemm(st, w.ob, DMODEL, m.o, qstream, qstream, nullptr, NTOK, DMODEL, DMODEL, 0);
}

static void run_fpn(hipStream_t st, const Fpn& f, float* x, Ws& w)
{
    launch_ln(st, x, f.ln_g, f.ln_b, w.xln);
    launch_gemm(st, w.xln, DMODEL, f.fc1, nullptr, w.mid, nullptr, NTOK, 2 * DMODEL, DMODEL, 1);
    launch_gemm(st, w.mid, 2 * DMODEL, f.fc2, x, x, nullptr, NTOK, DMODEL, 2 * DMODEL, 0);
}

extern "C" void kernel_launch(void* const* d_in, const int* in_sizes, int n_in,
                              void* d_out, int out_size, void* d_ws, size_t ws_size,
                              hipStream_t stream)
{
    (void)in_sizes; (void)n_in; (void)out_size; (void)ws_size;

    // Input flattening: jax pytree order (dict keys sorted alphabetically).
    // top {params,x,y}; params {cls,dec,enc,embed_x,embed_y};
    // linear {w,b} -> (b,w); mha -> (k, ln_b, ln_g, o, q, v);
    // fpn -> (fc1, fc2, ln_b, ln_g); dec layer -> (cross, fpn, self);
    // enc layer -> (fpn, mh)
    int idx = 0;
    auto F = [&]() { return (const float*)d_in[idx++]; };
    auto readLin = [&](Lin& l) { l.b = F(); l.w = F(); };
    auto readMha = [&](Mha& m) {
        readLin(m.k); m.ln_b = F(); m.ln_g = F();
        readLin(m.o); readLin(m.q); readLin(m.v);
    };
    auto readFpn = [&](Fpn& f) {
        readLin(f.fc1); readLin(f.fc2); f.ln_b = F(); f.ln_g = F();
    };

    Lin cls;            readLin(cls);
    Mha dec_self[3], dec_cross[3]; Fpn dec_fpn[3];
    for (int i = 0; i < 3; ++i) {
        readMha(dec_cross[i]); readFpn(dec_fpn[i]); readMha(dec_self[i]);
    }
    Mha enc_mh[3]; Fpn enc_fpn[3];
    for (int i = 0; i < 3; ++i) { readFpn(enc_fpn[i]); readMha(enc_mh[i]); }
    const float* embed_x = F();
    const float* embed_y = F();
    const int* x = (const int*)d_in[idx++];
    const int* y = (const int*)d_in[idx++];

    // Workspace carve.
    char* p = (char*)d_ws;
    auto carve = [&](size_t bytes) {
        void* r = (void*)p;
        p += (bytes + 255) & ~(size_t)255;
        return r;
    };
    Ws w;
    w.h   = (float*)carve((size_t)NTOK * DMODEL * 4);
    w.d   = (float*)carve((size_t)NTOK * DMODEL * 4);
    w.xln = (float*)carve((size_t)NTOK * DMODEL * 4);
    w.ob  = (float*)carve((size_t)NTOK * DMODEL * 4);
    w.mid = (float*)carve((size_t)NTOK * 2 * DMODEL * 4);
    w.qb  = (__bf16*)carve((size_t)NTOK * DMODEL * 2);
    w.kb  = (__bf16*)carve((size_t)NTOK * DMODEL * 2);
    w.vb  = (__bf16*)carve((size_t)NTOK * DMODEL * 2);

    // Forward.
    embed_pe_kernel<<<NTOK, DMODEL, 0, stream>>>(x, embed_x, w.h);
    embed_pe_kernel<<<NTOK, DMODEL, 0, stream>>>(y, embed_y, w.d);

    for (int l = 0; l < 3; ++l) {
        run_mha(stream, enc_mh[l], w.h, w.h, /*causal=*/0, w);
        run_fpn(stream, enc_fpn[l], w.h, w);
    }
    for (int l = 0; l < 3; ++l) {
        run_mha(stream, dec_self[l],  w.d, w.d, /*causal=*/1, w);
        run_mha(stream, dec_cross[l], w.d, w.h, /*causal=*/0, w);
        run_fpn(stream, dec_fpn[l], w.d, w);
    }
    launch_gemm(stream, w.d, DMODEL, cls, nullptr, (float*)d_out, nullptr,
                NTOK, 1000, DMODEL, 0);
}
